// Attention_15753940042128
// MI455X (gfx1250) — matrix operand order repaired and checked by reading, not compiled, not run on hardware
//
#include <hip/hip_runtime.h>
#include <hip/hip_bf16.h>

// ---------------- problem constants ----------------
#define BATCH 16
#define NTOK 577
#define CDIM 768
#define HEADS 12
#define HD 64
#define NPAD 608           // keys padded to 19 chunks of 32
#define QROWS 592          // queries padded to 37 tiles of 16
#define QKVC (3 * CDIM)    // 2304

typedef __attribute__((ext_vector_type(16))) __bf16 v16bf;
typedef __attribute__((ext_vector_type(8)))  __bf16 v8bf;
typedef __attribute__((ext_vector_type(8)))  float  v8f;

// scalar fp32 -> bf16 (native v_cvt; fptrunc = RTNE)
__device__ __forceinline__ __bf16 f2bf(float f) { return (__bf16)f; }

__device__ __forceinline__ v16bf cat8(v8bf lo, v8bf hi) {
  return __builtin_shufflevector(lo, hi, 0,1,2,3,4,5,6,7,8,9,10,11,12,13,14,15);
}

__device__ __forceinline__ v8f wmma_bf16(v16bf a, v16bf b, v8f c) {
  // D = A(16x32 bf16) * B(32x16 bf16) + C(16x16 f32)
  return __builtin_amdgcn_wmma_f32_16x16x32_bf16(false, a, false, b, (short)0, c,
                                                 false, false);
}

// =====================================================================
// Prep 1: elementwise fp32 -> bf16 (vector x8, packed converts)
// =====================================================================
__global__ void __launch_bounds__(256)
cvt_f32_bf16(const float* __restrict__ src, __bf16* __restrict__ dst, int n8) {
  int i = blockIdx.x * 256 + threadIdx.x;
  if (i < n8) {
    v8f s = *(const v8f*)(src + (size_t)i * 8);
    *(v8bf*)(dst + (size_t)i * 8) = __builtin_convertvector(s, v8bf);
  }
}

// =====================================================================
// Prep 2: pad rel_pos_bias [12,577,577] -> [12,592,608] with -1e30.
// The pad region doubles as the softmax key mask (no guards in hot loop).
// =====================================================================
__global__ void __launch_bounds__(256)
pad_bias_kernel(const float* __restrict__ relb, float* __restrict__ relbPad) {
  const int h = blockIdx.y;
  const int flat = blockIdx.x * 256 + threadIdx.x;   // < 592*608
  const int row = flat / NPAD;
  const int col = flat - row * NPAD;
  float v = (row < NTOK && col < NTOK)
                ? relb[((size_t)h * NTOK + row) * NTOK + col]
                : -1e30f;
  relbPad[(size_t)h * (QROWS * NPAD) + flat] = v;
}

// =====================================================================
// Kernel 1: QKV GEMM (all-bf16 operands).  Y = X @ Wqkv^T + bias ->
//   Q [B,H,NPAD,64] bf16 (pre-scaled by hd^-0.5)
//   K [B,H,NPAD,64] bf16
//   Vt[B,H,64,NPAD] bf16
// Each wave: 16(M) x 64(N); per k-step: 4 B-loads batched (one clause),
// next A-frag prefetched under the 4 WMMAs.
// Block = 128 (4 waves = 256 cols). Grid = (577, 2304/256 = 9).
// =====================================================================
__global__ void __launch_bounds__(128)
qkv_kernel(const __bf16* __restrict__ xbf, const __bf16* __restrict__ wbf,
           const float* __restrict__ q_bias, const float* __restrict__ v_bias,
           __bf16* __restrict__ Q, __bf16* __restrict__ K,
           __bf16* __restrict__ Vt) {
  const int lane = threadIdx.x & 31;
  const int wave = threadIdx.x >> 5;
  const int half = lane >> 4;
  const int l15  = lane & 15;

  const int m0 = blockIdx.x * 16;               // 9232 = 577*16 rows exactly
  const int n0 = blockIdx.y * 256 + wave * 64;  // output col group

  const __bf16* ap  = xbf + (size_t)(m0 + l15) * CDIM + half * 16;
  const __bf16* bp0 = wbf + (size_t)(n0 + l15) * CDIM + half * 16;

  v8f acc[4];
#pragma unroll
  for (int i = 0; i < 4; ++i) acc[i] = (v8f){0.f,0.f,0.f,0.f,0.f,0.f,0.f,0.f};

  // A software-pipelined one step ahead
  v8bf alo = *(const v8bf*)(ap);
  v8bf ahi = *(const v8bf*)(ap + 8);

  for (int k0 = 0; k0 < CDIM; k0 += 32) {
    v16bf a = cat8(alo, ahi);

    v16bf b[4];
#pragma unroll
    for (int i = 0; i < 4; ++i)
      b[i] = *(const v16bf*)(bp0 + (size_t)i * 16 * CDIM + k0);

    if (k0 + 32 < CDIM) {
      alo = *(const v8bf*)(ap + k0 + 32);
      ahi = *(const v8bf*)(ap + k0 + 40);
    }

#pragma unroll
    for (int i = 0; i < 4; ++i) acc[i] = wmma_bf16(a, b[i], acc[i]);
  }

  // ---- epilogue: uniform scalar routing, incremental div by 577
  const int b0 = m0 / NTOK;          // wave-uniform -> scalar
  const int t0 = m0 - b0 * NTOK;

#pragma unroll
  for (int i = 0; i < 4; ++i) {
    const int ntile = n0 + i * 16;          // uniform
    const int sec   = ntile / CDIM;         // uniform: 0=q 1=k 2=v
    const int cnb   = ntile - sec * CDIM;   // uniform
    const int h     = cnb / HD;             // uniform (16-col tile within head)
    const int d     = (cnb - h * HD) + l15;
    const float bias = (sec == 0) ? q_bias[cnb + l15]
                                  : ((sec == 2) ? v_bias[cnb + l15] : 0.f);
#pragma unroll
    for (int r = 0; r < 8; ++r) {
      int tt = t0 + r + half * 8;
      int bb = b0;
      if (tt >= NTOK) { tt -= NTOK; bb += 1; }
      const int bh = bb * HEADS + h;
      const float v = acc[i][r] + bias;
      if (sec == 0) {
        Q[((size_t)bh * NPAD + tt) * HD + d] = f2bf(v * 0.125f);
      } else if (sec == 1) {
        K[((size_t)bh * NPAD + tt) * HD + d] = f2bf(v);
      } else {
        Vt[((size_t)bh * HD + d) * NPAD + tt] = f2bf(v);
      }
    }
  }
}

// =====================================================================
// Kernel 2: flash attention per (b, h, 16-query tile). One wave/block.
//   S = Q K^T + relbPad (pad = -1e30 acts as mask), online softmax,
//   O += P V, O /= l.  8 bf16 WMMAs per 32-key chunk.
// =====================================================================
__global__ void __launch_bounds__(32)
attn_kernel(const __bf16* __restrict__ Q, const __bf16* __restrict__ K,
            const __bf16* __restrict__ Vt, const float* __restrict__ relbPad,
            __bf16* __restrict__ O) {
  __shared__ __attribute__((aligned(64))) __bf16 Plds[16 * 32];

  const int lane = threadIdx.x & 31;
  const int half = lane >> 4;
  const int l15  = lane & 15;

  const int q0 = blockIdx.x * 16;
  const int h  = blockIdx.y;
  const int b  = blockIdx.z;
  const int bh = b * HEADS + h;

  // Q A-frags (hd=64 -> two 16x32 chunks), loaded once
  v16bf aq[2];
  {
    const __bf16* qp = Q + ((size_t)bh * NPAD + q0 + l15) * HD;
#pragma unroll
    for (int c = 0; c < 2; ++c) {
      aq[c] = cat8(*(const v8bf*)(qp + c * 32 + half * 16),
                   *(const v8bf*)(qp + c * 32 + half * 16 + 8));
    }
  }

  v8f o0 = {0.f,0.f,0.f,0.f,0.f,0.f,0.f,0.f};
  v8f o1 = o0, o2 = o0, o3 = o0;
  float mrow[8], lrow[8];
#pragma unroll
  for (int r = 0; r < 8; ++r) { mrow[r] = -1e30f; lrow[r] = 0.f; }

  // padded bias base for this wave's row block
  const float* bbias =
      relbPad + ((size_t)h * QROWS + q0 + half * 8) * NPAD + l15;

  for (int j = 0; j < NPAD; j += 32) {
    // ---- S tiles for keys [j,j+16) and [j+16,j+32): batch 4 K-frag loads
    const __bf16* kbase = K + ((size_t)bh * NPAD + j + l15) * HD + half * 16;
    v16bf kf[4];
#pragma unroll
    for (int c = 0; c < 2; ++c) {
      kf[c]     = *(const v16bf*)(kbase + c * 32);
      kf[2 + c] = *(const v16bf*)(kbase + (size_t)16 * HD + c * 32);
    }
    v8f s0 = {0.f,0.f,0.f,0.f,0.f,0.f,0.f,0.f};
    v8f s1 = s0;
    s0 = wmma_bf16(aq[0], kf[0], s0);
    s0 = wmma_bf16(aq[1], kf[1], s0);
    s1 = wmma_bf16(aq[0], kf[2], s1);
    s1 = wmma_bf16(aq[1], kf[3], s1);

    __syncthreads();   // protect LDS tile from previous iteration's readers
#pragma unroll
    for (int r = 0; r < 8; ++r) {
      const float* bp = bbias + (size_t)r * NPAD + j;
      float v0 = s0[r] + bp[0];    // pad rows/cols carry -1e30 -> masked
      float v1 = s1[r] + bp[16];

      float cm = fmaxf(v0, v1);
      cm = fmaxf(cm, __shfl_xor(cm, 1, 32));
      cm = fmaxf(cm, __shfl_xor(cm, 2, 32));
      cm = fmaxf(cm, __shfl_xor(cm, 4, 32));
      cm = fmaxf(cm, __shfl_xor(cm, 8, 32));

      float mnew = fmaxf(mrow[r], cm);
      float corr = __expf(mrow[r] - mnew);
      float p0 = __expf(v0 - mnew);
      float p1 = __expf(v1 - mnew);
      float rs = p0 + p1;
      rs += __shfl_xor(rs, 1, 32);
      rs += __shfl_xor(rs, 2, 32);
      rs += __shfl_xor(rs, 4, 32);
      rs += __shfl_xor(rs, 8, 32);

      lrow[r] = lrow[r] * corr + rs;
      mrow[r] = mnew;
      o0[r] *= corr; o1[r] *= corr; o2[r] *= corr; o3[r] *= corr;

      Plds[(r + half * 8) * 32 + l15]      = f2bf(p0);
      Plds[(r + half * 8) * 32 + 16 + l15] = f2bf(p1);
    }
    __syncthreads();

    // ---- reread P in A-frag layout
    const __bf16* pp = Plds + l15 * 32;
    v16bf apf = cat8(*(const v8bf*)(pp + half * 16),
                     *(const v8bf*)(pp + half * 16 + 8));

    // ---- O += P * V  (Vt B-frags contiguous 32B per lane, batched)
    const __bf16* vp = Vt + ((size_t)bh * HD + l15) * NPAD + j + half * 16;
    v16bf vf[4];
#pragma unroll
    for (int i = 0; i < 4; ++i)
      vf[i] = *(const v16bf*)(vp + (size_t)i * 16 * NPAD);
    o0 = wmma_bf16(apf, vf[0], o0);
    o1 = wmma_bf16(apf, vf[1], o1);
    o2 = wmma_bf16(apf, vf[2], o2);
    o3 = wmma_bf16(apf, vf[3], o3);
  }

  // ---- finalize & store O[b, tok, h*64 + d] (bf16)
#pragma unroll
  for (int r = 0; r < 8; ++r) {
    int tok = q0 + r + half * 8;
    if (tok < NTOK) {
      float inv = 1.0f / fmaxf(lrow[r], 1e-20f);
      __bf16* op = O + ((size_t)b * NTOK + tok) * CDIM + h * HD;
      op[0 * 16 + l15] = f2bf(o0[r] * inv);
      op[1 * 16 + l15] = f2bf(o1[r] * inv);
      op[2 * 16 + l15] = f2bf(o2[r] * inv);
      op[3 * 16 + l15] = f2bf(o3[r] * inv);
    }
  }
}

// =====================================================================
// Kernel 3: out = O @ Wp^T + bp (bf16 operands, fp32 out).
// Each wave: 16x64, batched B loads + pipelined A. Grid = (577, 3).
// =====================================================================
__global__ void __launch_bounds__(128)
proj_kernel(const __bf16* __restrict__ O, const __bf16* __restrict__ pwbf,
            const float* __restrict__ pb, float* __restrict__ out) {
  const int lane = threadIdx.x & 31;
  const int wave = threadIdx.x >> 5;
  const int half = lane >> 4;
  const int l15  = lane & 15;

  const int m0 = blockIdx.x * 16;
  const int n0 = blockIdx.y * 256 + wave * 64;

  const __bf16* ap  = O + (size_t)(m0 + l15) * CDIM + half * 16;
  const __bf16* bp0 = pwbf + (size_t)(n0 + l15) * CDIM + half * 16;

  v8f acc[4];
#pragma unroll
  for (int i = 0; i < 4; ++i) acc[i] = (v8f){0.f,0.f,0.f,0.f,0.f,0.f,0.f,0.f};

  v8bf alo = *(const v8bf*)(ap);
  v8bf ahi = *(const v8bf*)(ap + 8);

  for (int k0 = 0; k0 < CDIM; k0 += 32) {
    v16bf a = cat8(alo, ahi);

    v16bf b[4];
#pragma unroll
    for (int i = 0; i < 4; ++i)
      b[i] = *(const v16bf*)(bp0 + (size_t)i * 16 * CDIM + k0);

    if (k0 + 32 < CDIM) {
      alo = *(const v8bf*)(ap + k0 + 32);
      ahi = *(const v8bf*)(ap + k0 + 40);
    }

#pragma unroll
    for (int i = 0; i < 4; ++i) acc[i] = wmma_bf16(a, b[i], acc[i]);
  }

#pragma unroll
  for (int i = 0; i < 4; ++i) {
    const int n = n0 + i * 16 + l15;
    const float bias = pb[n];
#pragma unroll
    for (int r = 0; r < 8; ++r) {
      int m = m0 + r + half * 8;   // < 9232 always
      out[(size_t)m * CDIM + n] = acc[i][r] + bias;
    }
  }
}

// =====================================================================
extern "C" void kernel_launch(void* const* d_in, const int* in_sizes, int n_in,
                              void* d_out, int out_size, void* d_ws,
                              size_t ws_size, hipStream_t stream) {
  (void)in_sizes; (void)n_in; (void)out_size; (void)ws_size;

  const float* x      = (const float*)d_in[0];  // [16,577,768]
  const float* relb   = (const float*)d_in[1];  // [12,577,577]
  const float* wqkv   = (const float*)d_in[2];  // [2304,768]
  const float* q_bias = (const float*)d_in[3];  // [768]
  const float* v_bias = (const float*)d_in[4];  // [768]
  const float* pw     = (const float*)d_in[5];  // [768,768]
  const float* pb     = (const float*)d_in[6];  // [768]
  float* out = (float*)d_out;

  auto al = [](size_t v) { return (v + 255) & ~(size_t)255; };
  char* ws = (char*)d_ws;
  size_t off = 0;

  const size_t nX  = (size_t)BATCH * NTOK * CDIM;        // 7,090,176
  const size_t nW  = (size_t)QKVC * CDIM;                // 1,769,472
  const size_t nPW = (size_t)CDIM * CDIM;                //   589,824
  const size_t nQKV = (size_t)BATCH * HEADS * NPAD * HD; // 7,471,104
  const size_t nRB = (size_t)HEADS * QROWS * NPAD;       // 4,319,232

  __bf16* xbf  = (__bf16*)(ws + off); off += al(nX * 2);
  __bf16* wbf  = (__bf16*)(ws + off); off += al(nW * 2);
  __bf16* pwbf = (__bf16*)(ws + off); off += al(nPW * 2);
  float* relbPad = (float*)(ws + off); off += al(nRB * 4);
  __bf16* Qb = (__bf16*)(ws + off); off += al(nQKV * 2);
  __bf16* Kb = (__bf16*)(ws + off); off += al(nQKV * 2);
  __bf16* Vt = (__bf16*)(ws + off); off += al(nQKV * 2);
  __bf16* Ob = (__bf16*)(ws + off); off += al(nX * 2);

  // ---- prep: one-time conversions + padded/masked bias
  cvt_f32_bf16<<<(int)(nX / 8 / 256), 256, 0, stream>>>(x, xbf, (int)(nX / 8));
  cvt_f32_bf16<<<(int)(nW / 8 / 256), 256, 0, stream>>>(wqkv, wbf, (int)(nW / 8));
  cvt_f32_bf16<<<(int)(nPW / 8 / 256), 256, 0, stream>>>(pw, pwbf, (int)(nPW / 8));
  pad_bias_kernel<<<dim3(QROWS * NPAD / 256, HEADS), 256, 0, stream>>>(relb, relbPad);

  // ---- 1) QKV GEMM + routing
  qkv_kernel<<<dim3(577, 9), 128, 0, stream>>>(xbf, wbf, q_bias, v_bias, Qb, Kb, Vt);

  // ---- 2) flash attention
  attn_kernel<<<dim3(37, HEADS, BATCH), 32, 0, stream>>>(Qb, Kb, Vt, relbPad, Ob);

  // ---- 3) projection GEMM
  proj_kernel<<<dim3(577, 3), 128, 0, stream>>>(Ob, pwbf, pb, out);
}
